// MultiHeadAttentionWithRoPE_51582557225796
// MI455X (gfx1250) — matrix operand order repaired and checked by reading, not compiled, not run on hardware
//
#include <hip/hip_runtime.h>
#include <hip/hip_bf16.h>

// ---- constants for this problem ----
#define BATCH   2
#define TSEQ    2048
#define EMBED   1024
#define HEADS   16
#define HDIM    64
#define MROWS   (BATCH*TSEQ)   // 4096

typedef __attribute__((ext_vector_type(16))) __bf16 v16bf;
typedef __attribute__((ext_vector_type(8)))  float  v8f;

union Frag16 { v16bf v; uint4 u[2]; };         // 16 bf16 = 32B = 8 VGPRs
union F8     { v8f  v;  float f[8]; };         // f32 accumulator tile

__device__ __forceinline__ unsigned short f2bf(float f) {
    unsigned int u = __float_as_uint(f);
    unsigned int r = (u + 0x7FFFu + ((u >> 16) & 1u)) >> 16;  // RNE
    return (unsigned short)r;
}

__device__ __forceinline__ v8f wmma_bf16(v16bf a, v16bf b, v8f c) {
    return __builtin_amdgcn_wmma_f32_16x16x32_bf16(
        false, a, false, b, (short)0, c, false, false);
}

// ---------------- elementwise converts ----------------
__global__ void cvt_f32_bf16(const float* __restrict__ in,
                             unsigned short* __restrict__ out, int n) {
    int i = blockIdx.x * 256 + threadIdx.x;
    if (i < n) out[i] = f2bf(in[i]);
}

// RoPE: read fp32 proj [M,1024] row-major, write bf16 [B*H, T, 64]
__global__ void rope_kernel(const float* __restrict__ in,
                            unsigned short* __restrict__ out) {
    int idx = blockIdx.x * 256 + threadIdx.x;      // B*T*H*32 = 2097152
    if (idx >= BATCH * TSEQ * HEADS * 32) return;
    int i = idx & 31;                 // rotary pair index 0..31
    int h = (idx >> 5) & (HEADS - 1);
    int t = (idx >> 9) & (TSEQ - 1);
    int b = idx >> 20;
    const float* row = in + (size_t)(b * TSEQ + t) * EMBED + h * HDIM;
    float x1 = row[i], x2 = row[i + 32];
    float invf = powf(10000.0f, -(float)i / 32.0f);
    float ang = (float)t * invf;
    float c = cosf(ang), s = sinf(ang);
    unsigned short* orow = out + (size_t)((b * HEADS + h) * TSEQ + t) * HDIM;
    orow[i]      = f2bf(x1 * c - x2 * s);
    orow[i + 32] = f2bf(x2 * c + x1 * s);
}

// V: fp32 [M,1024] -> bf16 [B*H, 64, T] (transposed so V is a contiguous B operand)
__global__ void vtrans_kernel(const float* __restrict__ in,
                              unsigned short* __restrict__ out) {
    int idx = blockIdx.x * 256 + threadIdx.x;      // B*T*1024
    if (idx >= MROWS * EMBED) return;
    int o = idx & (EMBED - 1);
    int m = idx >> 10;
    int h = o >> 6, d = o & 63;
    int b = m >> 11, t = m & (TSEQ - 1);
    out[(size_t)((b * HEADS + h) * HDIM + d) * TSEQ + t] = f2bf(in[idx]);
}

// ---------------- bf16 WMMA GEMM:  Out[M,N] = A[M,K] * W[N,K]^T + bias ----------------
// one wave (32 threads) per 32x32 output tile
__global__ __launch_bounds__(32) void gemm_bf16(
    const unsigned short* __restrict__ A,
    const unsigned short* __restrict__ W,
    const float* __restrict__ bias,
    float* __restrict__ Out,
    int M, int N, int K)
{
    const int lane = threadIdx.x;
    const int lo = lane & 15, hi = lane >> 4;
    const int ntiles = N >> 5;
    const int mt = blockIdx.x / ntiles;
    const int nt = blockIdx.x % ntiles;
    const int m0 = mt * 32, n0 = nt * 32;

    F8 acc[2][2];
#pragma unroll
    for (int a = 0; a < 2; ++a)
#pragma unroll
        for (int b = 0; b < 2; ++b)
#pragma unroll
            for (int r = 0; r < 8; ++r) acc[a][b].f[r] = 0.0f;

    const unsigned short* arow[2] = { A + (size_t)(m0 + lo) * K,
                                      A + (size_t)(m0 + 16 + lo) * K };
    const unsigned short* wrow[2] = { W + (size_t)(n0 + lo) * K,
                                      W + (size_t)(n0 + 16 + lo) * K };

    for (int k0 = 0; k0 < K; k0 += 32) {
        Frag16 af[2], bf[2];
#pragma unroll
        for (int rt = 0; rt < 2; ++rt) {
            const unsigned short* p = arow[rt] + k0 + hi * 16;  // A-operand layout
            af[rt].u[0] = *(const uint4*)(p);
            af[rt].u[1] = *(const uint4*)(p + 8);
        }
#pragma unroll
        for (int ct = 0; ct < 2; ++ct) {
            const unsigned short* p = wrow[ct] + k0 + hi * 16;  // B-operand layout
            bf[ct].u[0] = *(const uint4*)(p);
            bf[ct].u[1] = *(const uint4*)(p + 8);
        }
#pragma unroll
        for (int rt = 0; rt < 2; ++rt)
#pragma unroll
            for (int ct = 0; ct < 2; ++ct)
                acc[rt][ct].v = wmma_bf16(af[rt].v, bf[ct].v, acc[rt][ct].v);
    }

    float bv[2] = { bias[n0 + lo], bias[n0 + 16 + lo] };
#pragma unroll
    for (int rt = 0; rt < 2; ++rt)
#pragma unroll
        for (int ct = 0; ct < 2; ++ct)
#pragma unroll
            for (int r = 0; r < 8; ++r) {
                size_t row = (size_t)(m0 + rt * 16 + hi * 8 + r);
                Out[row * N + n0 + ct * 16 + lo] = acc[rt][ct].f[r] + bv[ct];
            }
}

// ---------------- flash attention: one wave per (bh, 32 query rows) ----------------
__global__ __launch_bounds__(32) void attn_kernel(
    const unsigned short* __restrict__ qbf,   // [BH, T, 64]
    const unsigned short* __restrict__ kbf,   // [BH, T, 64]
    const unsigned short* __restrict__ vT,    // [BH, 64, T]
    unsigned short* __restrict__ out)         // [B*T, 1024] bf16
{
    __shared__ float sS[32][33];
    __shared__ unsigned short sP[32][32];
    __shared__ float sAlpha[32];
    __shared__ float sL[32];

    const int lane = threadIdx.x;
    const int lo = lane & 15, hi = lane >> 4;
    const int bh = blockIdx.y;
    const int b = bh >> 4, h = bh & 15;
    const int q0 = blockIdx.x * 32;

    const unsigned short* qb = qbf + (size_t)bh * TSEQ * HDIM;
    const unsigned short* kb = kbf + (size_t)bh * TSEQ * HDIM;
    const unsigned short* vb = vT  + (size_t)bh * HDIM * TSEQ;

    // Q fragments, resident for whole kernel: rt = row tile, kc = head-dim chunk
    Frag16 qf[2][2];
#pragma unroll
    for (int rt = 0; rt < 2; ++rt) {
        const unsigned short* rp = qb + (size_t)(q0 + rt * 16 + lo) * HDIM;
#pragma unroll
        for (int kc = 0; kc < 2; ++kc) {
            const unsigned short* p = rp + kc * 32 + hi * 16;
            qf[rt][kc].u[0] = *(const uint4*)(p);
            qf[rt][kc].u[1] = *(const uint4*)(p + 8);
        }
    }

    F8 Oacc[2][4];
#pragma unroll
    for (int rt = 0; rt < 2; ++rt)
#pragma unroll
        for (int dt = 0; dt < 4; ++dt)
#pragma unroll
            for (int r = 0; r < 8; ++r) Oacc[rt][dt].f[r] = 0.0f;

    float m_run = -1e30f, l_run = 0.0f;
    const float scale = 0.125f;   // 1/sqrt(64)

    for (int s0 = 0; s0 < TSEQ; s0 += 32) {
        // ---- scores S = Q * K^T for 32 keys ----
        F8 acc[2][2];
#pragma unroll
        for (int a = 0; a < 2; ++a)
#pragma unroll
            for (int c = 0; c < 2; ++c)
#pragma unroll
                for (int r = 0; r < 8; ++r) acc[a][c].f[r] = 0.0f;

#pragma unroll
        for (int st = 0; st < 2; ++st) {
            const unsigned short* rp = kb + (size_t)(s0 + st * 16 + lo) * HDIM;
#pragma unroll
            for (int kc = 0; kc < 2; ++kc) {
                Frag16 kf;
                const unsigned short* p = rp + kc * 32 + hi * 16;
                kf.u[0] = *(const uint4*)(p);
                kf.u[1] = *(const uint4*)(p + 8);
#pragma unroll
                for (int rt = 0; rt < 2; ++rt)
                    acc[rt][st].v = wmma_bf16(qf[rt][kc].v, kf.v, acc[rt][st].v);
            }
        }

        // ---- spill scaled scores to LDS (C/D layout -> [row][col]) ----
#pragma unroll
        for (int rt = 0; rt < 2; ++rt)
#pragma unroll
            for (int st = 0; st < 2; ++st)
#pragma unroll
                for (int r = 0; r < 8; ++r)
                    sS[rt * 16 + hi * 8 + r][st * 16 + lo] = acc[rt][st].f[r] * scale;
        __syncthreads();

        // ---- online softmax: lane owns row==lane ----
        float rowmax = m_run;
#pragma unroll
        for (int j = 0; j < 32; ++j) rowmax = fmaxf(rowmax, sS[lane][j]);
        float alpha = __expf(m_run - rowmax);
        float psum = 0.0f;
#pragma unroll
        for (int j = 0; j < 32; ++j) {
            float p = __expf(sS[lane][j] - rowmax);
            psum += p;
            sP[lane][j] = f2bf(p);
        }
        l_run = l_run * alpha + psum;
        m_run = rowmax;
        sAlpha[lane] = alpha;
        __syncthreads();

        // ---- rescale O accumulators ----
#pragma unroll
        for (int rt = 0; rt < 2; ++rt)
#pragma unroll
            for (int r = 0; r < 8; ++r) {
                float a = sAlpha[rt * 16 + hi * 8 + r];
#pragma unroll
                for (int dt = 0; dt < 4; ++dt) Oacc[rt][dt].f[r] *= a;
            }

        // ---- O += P * V : P from LDS in A-operand layout, V from vT contiguous ----
        Frag16 pf[2];
#pragma unroll
        for (int rt = 0; rt < 2; ++rt) {
            const unsigned short* pr = &sP[rt * 16 + lo][0] + hi * 16;
            pf[rt].u[0] = *(const uint4*)(pr);
            pf[rt].u[1] = *(const uint4*)(pr + 8);
        }
#pragma unroll
        for (int dt = 0; dt < 4; ++dt) {
            const unsigned short* vr = vb + (size_t)(dt * 16 + lo) * TSEQ + s0 + hi * 16;
            Frag16 vf;
            vf.u[0] = *(const uint4*)(vr);
            vf.u[1] = *(const uint4*)(vr + 8);
#pragma unroll
            for (int rt = 0; rt < 2; ++rt)
                Oacc[rt][dt].v = wmma_bf16(pf[rt].v, vf.v, Oacc[rt][dt].v);
        }
        __syncthreads();   // protect LDS reuse next iteration
    }

    sL[lane] = l_run;
    __syncthreads();

#pragma unroll
    for (int rt = 0; rt < 2; ++rt)
#pragma unroll
        for (int r = 0; r < 8; ++r) {
            int rowIdx = rt * 16 + hi * 8 + r;
            float linv = 1.0f / sL[rowIdx];
            int t = q0 + rowIdx;
            size_t orow = ((size_t)(b * TSEQ + t)) * EMBED + h * HDIM;
#pragma unroll
            for (int dt = 0; dt < 4; ++dt)
                out[orow + dt * 16 + lo] = f2bf(Oacc[rt][dt].f[r] * linv);
        }
}

// ---------------- host-side launch ----------------
extern "C" void kernel_launch(void* const* d_in, const int* in_sizes, int n_in,
                              void* d_out, int out_size, void* d_ws, size_t ws_size,
                              hipStream_t stream) {
    (void)in_sizes; (void)n_in; (void)out_size; (void)ws_size;
    const float* x  = (const float*)d_in[0];
    const float* Wq = (const float*)d_in[1];
    const float* bq = (const float*)d_in[2];
    const float* Wk = (const float*)d_in[3];
    const float* bk = (const float*)d_in[4];
    const float* Wv = (const float*)d_in[5];
    const float* bv = (const float*)d_in[6];
    const float* Wo = (const float*)d_in[7];
    const float* bo = (const float*)d_in[8];
    float* out = (float*)d_out;

    char* ws = (char*)d_ws;
    unsigned short* xbf  = (unsigned short*)(ws);                     //  8 MB
    unsigned short* wqb  = (unsigned short*)(ws + (size_t)8  * 1048576);
    unsigned short* wkb  = (unsigned short*)(ws + (size_t)10 * 1048576);
    unsigned short* wvb  = (unsigned short*)(ws + (size_t)12 * 1048576);
    unsigned short* wob  = (unsigned short*)(ws + (size_t)14 * 1048576);
    float*          f32s = (float*)         (ws + (size_t)16 * 1048576); // 16 MB
    unsigned short* qbf  = (unsigned short*)(ws + (size_t)32 * 1048576);
    unsigned short* kbf  = (unsigned short*)(ws + (size_t)40 * 1048576);
    unsigned short* vT   = (unsigned short*)(ws + (size_t)48 * 1048576);
    unsigned short* abf  = (unsigned short*)(ws + (size_t)56 * 1048576);

    const int NX = MROWS * EMBED;       // 4 Mi elements
    const int NW = EMBED * EMBED;       // 1 Mi elements
    cvt_f32_bf16<<<(NX + 255) / 256, 256, 0, stream>>>(x,  xbf, NX);
    cvt_f32_bf16<<<(NW + 255) / 256, 256, 0, stream>>>(Wq, wqb, NW);
    cvt_f32_bf16<<<(NW + 255) / 256, 256, 0, stream>>>(Wk, wkb, NW);
    cvt_f32_bf16<<<(NW + 255) / 256, 256, 0, stream>>>(Wv, wvb, NW);
    cvt_f32_bf16<<<(NW + 255) / 256, 256, 0, stream>>>(Wo, wob, NW);

    dim3 gg((MROWS / 32) * (EMBED / 32));          // 4096 waves
    const int NR = BATCH * TSEQ * HEADS * 32;      // rope threads

    gemm_bf16<<<gg, 32, 0, stream>>>(xbf, wqb, bq, f32s, MROWS, EMBED, EMBED);
    rope_kernel<<<(NR + 255) / 256, 256, 0, stream>>>(f32s, qbf);

    gemm_bf16<<<gg, 32, 0, stream>>>(xbf, wkb, bk, f32s, MROWS, EMBED, EMBED);
    rope_kernel<<<(NR + 255) / 256, 256, 0, stream>>>(f32s, kbf);

    gemm_bf16<<<gg, 32, 0, stream>>>(xbf, wvb, bv, f32s, MROWS, EMBED, EMBED);
    vtrans_kernel<<<(NX + 255) / 256, 256, 0, stream>>>(f32s, vT);

    attn_kernel<<<dim3(TSEQ / 32, BATCH * HEADS), 32, 0, stream>>>(qbf, kbf, vT, abf);

    gemm_bf16<<<gg, 32, 0, stream>>>(abf, wob, bo, out, MROWS, EMBED, EMBED);
}
